// ContrastiveLoss_27376121545224
// MI455X (gfx1250) — matrix operand order repaired and checked by reading, not compiled, not run on hardware
//
#include <hip/hip_runtime.h>

// ---------------------------------------------------------------------------
// ContrastiveLoss on MI455X (gfx1250, wave32, WMMA)
//
// ys [4096,2048] f32, labels [4096] i64 -> scalar f32
//   yn  = ys / max(||ys||_2, 1e-6)       (rows normalized, stored f16 in ws)
//   cos = yn @ yn^T                       (f16 WMMA, f32 accumulate)
//   loss(i>j) = same ? relu(2-cos)^2 : cos^2 ; mean over b(b-1)/2 pairs
//
// Compute-bound SYRK (~34 GFLOP over lower-tri tiles) -> f16 WMMA.
// 128x128 block tile, 8 waves, wave = 32x64 sub-tile = 8 WMMA/chunk.
// LDS double-buffered: one barrier per K chunk; next chunk's global loads
// issue before the WMMA burst so latency hides under the matrix pipe.
// yn (16MB) is L2-resident (192MB), so cross-block B reuse is cheap.
// ---------------------------------------------------------------------------

typedef __attribute__((ext_vector_type(16))) _Float16 v16h;
typedef __attribute__((ext_vector_type(8)))  _Float16 v8h;
typedef __attribute__((ext_vector_type(4)))  _Float16 v4h;
typedef __attribute__((ext_vector_type(8)))  float    v8f;

#define BSZ    4096          // batch (rows)
#define KDIM   2048          // feature dim
#define EPSN   1e-6f
#define MARGIN 2.0f

#define TILE   128           // block tile (128x128 output)
#define KC     32            // K chunk per WMMA step
#define NKT    (KDIM / KC)   // 64 K iterations
#define LDT    40            // padded LDS row stride in halves (80B = 5*16B)

// ---------------------------------------------------------------------------
// Kernel 1: per-row L2 normalize, f32 -> f16
// ---------------------------------------------------------------------------
__global__ __launch_bounds__(256) void k_normalize(const float* __restrict__ ys,
                                                   _Float16* __restrict__ yn) {
  const int row = blockIdx.x;
  const int tid = threadIdx.x;
  const float4* src = (const float4*)(ys + (size_t)row * KDIM);

  float4 a = src[tid];
  float4 b = src[tid + 256];
  float ss = a.x*a.x + a.y*a.y + a.z*a.z + a.w*a.w
           + b.x*b.x + b.y*b.y + b.z*b.z + b.w*b.w;

  __shared__ float red[256];
  red[tid] = ss;
  __syncthreads();
  for (int s = 128; s > 0; s >>= 1) {
    if (tid < s) red[tid] += red[tid + s];
    __syncthreads();
  }
  const float sc = 1.0f / fmaxf(sqrtf(red[0]), EPSN);

  v4h o0 = { (_Float16)(a.x*sc), (_Float16)(a.y*sc),
             (_Float16)(a.z*sc), (_Float16)(a.w*sc) };
  v4h o1 = { (_Float16)(b.x*sc), (_Float16)(b.y*sc),
             (_Float16)(b.z*sc), (_Float16)(b.w*sc) };
  v4h* dst = (v4h*)(yn + (size_t)row * KDIM);
  dst[tid]       = o0;
  dst[tid + 256] = o1;
}

// ---------------------------------------------------------------------------
// Kernel 2: lower-triangular tiled Gram + pair loss.
// Grid = T(T+1)/2 blocks (T = 32), 256 threads = 8 waves.
// Wave (wr=w>>1, wc=w&1) owns rows [i0+32wr, +32) x cols [j0+64wc, +64):
// 2 (ti) x 4 (nt) = 8 WMMA accumulators.
// ---------------------------------------------------------------------------
__global__ __launch_bounds__(256) void k_gram_loss(const _Float16* __restrict__ yn,
                                                   const long long* __restrict__ labels,
                                                   float* __restrict__ partials) {
  __shared__ _Float16 As[2][TILE * LDT];
  __shared__ _Float16 Bs[2][TILE * LDT];
  __shared__ int Li[TILE];
  __shared__ int Lj[TILE];
  __shared__ float red[256];

  const int tid  = threadIdx.x;
  const int wave = tid >> 5;
  const int lane = tid & 31;
  const int n16  = lane & 15;   // N (and A-row M) index within 16
  const int hi   = lane >> 4;   // lane half-group
  const int wr   = wave >> 1;   // wave row strip  (0..3) -> 32 rows
  const int wc   = wave & 1;    // wave col half   (0..1) -> 64 cols

  // linear block id -> (bi, bj) with bi >= bj (lower tri incl. diagonal)
  int id = blockIdx.x;
  int bi = (int)((sqrtf(8.0f * (float)id + 1.0f) - 1.0f) * 0.5f);
  while ((bi + 1) * (bi + 2) / 2 <= id) ++bi;
  while (bi * (bi + 1) / 2 > id) --bi;
  const int bj = id - bi * (bi + 1) / 2;
  const int i0 = bi * TILE;
  const int j0 = bj * TILE;

  // stage labels as i32 (covered by the preload barrier)
  if (tid < TILE) {
    Li[tid] = (int)labels[i0 + tid];
    Lj[tid] = (int)labels[j0 + tid];
  }

  // per-thread staging geometry: 512 v8h per matrix per chunk, 2 per thread
  const int r0 = (tid)        >> 2;                 // vectors 0..255
  const int c0 = ((tid) & 3)  * 8;
  const int r1 = (tid + 256)  >> 2;                 // vectors 256..511
  const int c1 = ((tid + 256) & 3) * 8;
  const _Float16* gA = yn + (size_t)i0 * KDIM;
  const _Float16* gB = yn + (size_t)j0 * KDIM;

  // ---- preload chunk 0 into buffer 0 ----
  {
    *(v8h*)(&As[0][r0 * LDT + c0]) = *(const v8h*)(gA + (size_t)r0 * KDIM + c0);
    *(v8h*)(&As[0][r1 * LDT + c1]) = *(const v8h*)(gA + (size_t)r1 * KDIM + c1);
    *(v8h*)(&Bs[0][r0 * LDT + c0]) = *(const v8h*)(gB + (size_t)r0 * KDIM + c0);
    *(v8h*)(&Bs[0][r1 * LDT + c1]) = *(const v8h*)(gB + (size_t)r1 * KDIM + c1);
  }
  __syncthreads();

  v8f acc[2][4] = {};   // [ti][nt]

  for (int kt = 0; kt < NKT; ++kt) {
    const int cur = kt & 1;
    const int nxt = cur ^ 1;
    const int kn  = (kt + 1) * KC;
    const bool more = (kt + 1 < NKT);

    // ---- issue next chunk's global loads early (latency under WMMAs) ----
    v8h sa0, sa1, sb0, sb1;
    if (more) {
      sa0 = *(const v8h*)(gA + (size_t)r0 * KDIM + kn + c0);
      sa1 = *(const v8h*)(gA + (size_t)r1 * KDIM + kn + c1);
      sb0 = *(const v8h*)(gB + (size_t)r0 * KDIM + kn + c0);
      sb1 = *(const v8h*)(gB + (size_t)r1 * KDIM + kn + c1);
      if (kt + 2 < NKT) {   // nudge chunk kt+2 toward L2/WGP$
        __builtin_prefetch(gA + (size_t)(tid >> 1) * KDIM + kn + KC, 0, 3);
        __builtin_prefetch(gB + (size_t)(tid >> 1) * KDIM + kn + KC, 0, 3);
      }
    }

    // ---- A fragments (ISA 16-bit A 16x32: h[0..7]=K(hi*8..), h[8..15]=K(16+hi*8..)) ----
    union { v16h v; v8h h[2]; } af[2];
#pragma unroll
    for (int ti = 0; ti < 2; ++ti) {
      const int arow = wr * 32 + ti * 16 + n16;
      af[ti].h[0] = *(const v8h*)(&As[cur][arow * LDT + hi * 8]);
      af[ti].h[1] = *(const v8h*)(&As[cur][arow * LDT + 16 + hi * 8]);
    }

    // ---- WMMA burst: 4 B frags x 2 A frags = 8 WMMAs ----
#pragma unroll
    for (int nt = 0; nt < 4; ++nt) {
      union { v16h v; v8h h[2]; } bf;   // B 32x16: col n16, h[0..7]=K(hi*8..), h[8..15]=K(16+hi*8..)
      const int brow = wc * 64 + nt * 16 + n16;
      bf.h[0] = *(const v8h*)(&Bs[cur][brow * LDT + hi * 8]);
      bf.h[1] = *(const v8h*)(&Bs[cur][brow * LDT + 16 + hi * 8]);
#pragma unroll
      for (int ti = 0; ti < 2; ++ti) {
        acc[ti][nt] = __builtin_amdgcn_wmma_f32_16x16x32_f16(
            false, af[ti].v, false, bf.v, (short)0, acc[ti][nt], false, false);
      }
    }

    // ---- stage next chunk into the other buffer ----
    if (more) {
      *(v8h*)(&As[nxt][r0 * LDT + c0]) = sa0;
      *(v8h*)(&As[nxt][r1 * LDT + c1]) = sa1;
      *(v8h*)(&Bs[nxt][r0 * LDT + c0]) = sb0;
      *(v8h*)(&Bs[nxt][r1 * LDT + c1]) = sb1;
    }
    __syncthreads();   // single barrier: nxt ready, cur free for reuse
  }

  // ---- pair loss from accumulators: D elem (vgpr v, lane): M=v+8*hi, N=n16 ----
  float part = 0.0f;
  const bool diag = (bi == bj);
#pragma unroll
  for (int ti = 0; ti < 2; ++ti) {
#pragma unroll
    for (int nt = 0; nt < 4; ++nt) {
      const int jn = wc * 64 + nt * 16 + n16;
      const int j  = j0 + jn;
      const int lj = Lj[jn];
#pragma unroll
      for (int v = 0; v < 8; ++v) {
        const int mi = wr * 32 + ti * 16 + v + 8 * hi;
        const int i  = i0 + mi;
        if (!diag || i > j) {
          const float c = acc[ti][nt][v];
          const float t = (Li[mi] == lj) ? fmaxf(MARGIN - c, 0.0f) : c;
          part += t * t;
        }
      }
    }
  }

  red[tid] = part;
  __syncthreads();
  for (int s = 128; s > 0; s >>= 1) {
    if (tid < s) red[tid] += red[tid + s];
    __syncthreads();
  }
  if (tid == 0) partials[blockIdx.x] = red[0];
}

// ---------------------------------------------------------------------------
// Kernel 3: deterministic final reduction -> mean
// ---------------------------------------------------------------------------
__global__ __launch_bounds__(256) void k_reduce(const float* __restrict__ partials,
                                                int n, float inv_npair,
                                                float* __restrict__ out) {
  __shared__ float red[256];
  const int tid = threadIdx.x;
  float s = 0.0f;
  for (int i = tid; i < n; i += 256) s += partials[i];
  red[tid] = s;
  __syncthreads();
  for (int st = 128; st > 0; st >>= 1) {
    if (tid < st) red[tid] += red[tid + st];
    __syncthreads();
  }
  if (tid == 0) out[0] = red[0] * inv_npair;
}

// ---------------------------------------------------------------------------
extern "C" void kernel_launch(void* const* d_in, const int* in_sizes, int n_in,
                              void* d_out, int out_size, void* d_ws, size_t ws_size,
                              hipStream_t stream) {
  (void)in_sizes; (void)n_in; (void)out_size; (void)ws_size;

  const float*     ys     = (const float*)d_in[0];
  const long long* labels = (const long long*)d_in[1];   // int64 labels
  float*           out    = (float*)d_out;

  _Float16* yn       = (_Float16*)d_ws;                                  // 16 MB
  float*    partials = (float*)((char*)d_ws + (size_t)BSZ * KDIM * sizeof(_Float16));

  const int T       = BSZ / TILE;           // 32 tiles per side
  const int nblocks = T * (T + 1) / 2;      // 528 lower-tri tile blocks

  k_normalize<<<BSZ, 256, 0, stream>>>(ys, yn);
  k_gram_loss<<<nblocks, 256, 0, stream>>>(yn, labels, partials);

  const float inv_npair = 1.0f / ((float)BSZ * (float)(BSZ - 1) * 0.5f);
  k_reduce<<<1, 256, 0, stream>>>(partials, nblocks, inv_npair, out);
}
